// BiLSTM_CRF_13889924235662
// MI455X (gfx1250) — matrix operand order repaired and checked by reading, not compiled, not run on hardware
//
#include <hip/hip_runtime.h>
#include <math.h>
#include <stdint.h>

#define SQ 512   // sequence length
#define BB 256   // batch
#define HH 100   // hidden
#define EE 100   // embed
#define TT 25    // tags
#define GP 448   // padded gate dim = 4*112
#define KP 128   // padded K for E/H (bf16 WMMA K multiple of 32)
#define KC 256   // padded K for concat(hf,hb) = 2*KP
#define NEMI 32  // padded tag dim

#if defined(__HIP_DEVICE_COMPILE__) && __has_builtin(__builtin_amdgcn_global_load_async_to_lds_b128)
#define HAVE_ASYNC_LDS 1
#else
#define HAVE_ASYNC_LDS 0
#endif

typedef __bf16 bf16;
typedef __attribute__((ext_vector_type(16))) __bf16 v16bf;
typedef __attribute__((ext_vector_type(8)))  float  v8f;

static __device__ __forceinline__ bf16 f2bf(float f) {
  union { float f; unsigned u; } in; in.f = f;
  unsigned u = in.u;
  unsigned short r = (unsigned short)((u + 0x7FFFu + ((u >> 16) & 1u)) >> 16);
  union { unsigned short s; bf16 h; } out; out.s = r;
  return out.h;
}

#if HAVE_ASYNC_LDS
typedef int v4i __attribute__((vector_size(4 * sizeof(int))));
typedef __attribute__((address_space(1))) v4i* gv4i_ptr;
typedef __attribute__((address_space(3))) v4i* lv4i_ptr;
static __device__ __forceinline__ void async_copy_b128(const void* gsrc, void* ldst) {
  __builtin_amdgcn_global_load_async_to_lds_b128(
      (gv4i_ptr)(uintptr_t)gsrc,
      (lv4i_ptr)(unsigned)(uintptr_t)ldst,
      0, 0);
}
static __device__ __forceinline__ void wait_async0() {
#if __has_builtin(__builtin_amdgcn_s_wait_asynccnt)
  __builtin_amdgcn_s_wait_asynccnt(0);
#else
  asm volatile("s_wait_asynccnt 0x0" ::: "memory");
#endif
}
#endif

// A fragment: 16x32 bf16, row-major source with 'stride' elements per row.
// lanes 0-15: M=lane, VGPR0-3 K=0..7, VGPR4-7 K=16..23
// lanes 16-31: M=lane-16, VGPR0-3 K=8..15, VGPR4-7 K=24..31
static __device__ __forceinline__ v16bf load_a_frag(const bf16* base, int stride,
                                                    int lane, int kbase) {
  int m = lane & 15, half = lane >> 4;
  const bf16* row = base + m * stride + kbase;
  v16bf a;
#pragma unroll
  for (int e = 0; e < 8; ++e)  a[e] = row[e + 8 * half];
#pragma unroll
  for (int e = 8; e < 16; ++e) a[e] = row[e + 8 + 8 * half];
  return a;
}

// B fragment: 32x16 bf16 (KxN), source laid out [N][K] row-major (stride elems).
// lanes 0-15: N=lane, K=0..7 then 16..23; lanes 16-31: K=8..15 then 24..31.
static __device__ __forceinline__ v16bf load_b_frag(const bf16* base, int stride,
                                                    int lane, int kbase) {
  int n = lane & 15, half = lane >> 4;
  const bf16* row = base + n * stride + kbase;
  v16bf b;
#pragma unroll
  for (int e = 0; e < 8; ++e)  b[e] = row[e + 8 * half];
#pragma unroll
  for (int e = 8; e < 16; ++e) b[e] = row[e + 8 + 8 * half];
  return b;
}

// ---------------- weight preparation (pad + bf16 convert) ----------------
__global__ void k_prep_wpad(const float* w_ih_f, const float* w_hh_f,
                            const float* w_ih_b, const float* w_hh_b,
                            bf16* whh_pad, bf16* wih_pad) {
  const int total = 2 * GP * KP;
  for (int id = blockIdx.x * blockDim.x + threadIdx.x; id < total;
       id += gridDim.x * blockDim.x) {
    int d = id / (GP * KP);
    int rem = id % (GP * KP);
    int n = rem / KP, k = rem % KP;
    int gate = n / 112, j = n % 112;
    float vh = 0.f, vi = 0.f;
    if (j < HH && k < HH) {
      const float* whh = d ? w_hh_b : w_hh_f;
      vh = whh[(gate * HH + j) * HH + k];
    }
    if (j < HH && k < EE) {
      const float* wih = d ? w_ih_b : w_ih_f;
      vi = wih[(gate * HH + j) * EE + k];
    }
    whh_pad[id] = f2bf(vh);
    wih_pad[id] = f2bf(vi);
  }
}

__global__ void k_prep_misc(const float* b_ih_f, const float* b_hh_f,
                            const float* b_ih_b, const float* b_hh_b,
                            const float* w_out, float* bias_sum, bf16* woT) {
  const int tid = blockIdx.x * blockDim.x + threadIdx.x;
  const int stride = gridDim.x * blockDim.x;
  for (int id = tid; id < 2 * GP; id += stride) {
    int d = id / GP, n = id % GP;
    int gate = n / 112, j = n % 112;
    float v = 0.f;
    if (j < HH)
      v = d ? (b_ih_b[gate * HH + j] + b_hh_b[gate * HH + j])
            : (b_ih_f[gate * HH + j] + b_hh_f[gate * HH + j]);
    bias_sum[id] = v;
  }
  for (int id = tid; id < NEMI * KC; id += stride) {
    int n = id / KC, k = id % KC;
    float v = 0.f;
    if (n < TT) {
      if (k < HH) v = w_out[n * (2 * HH) + k];
      else if (k >= KP && k < KP + HH) v = w_out[n * (2 * HH) + HH + (k - KP)];
    }
    woT[id] = f2bf(v);
  }
}

// ---------------- embedding gather -> padded bf16 [S][B][KP] ----------------
__global__ void k_embed(const int* sentence, const float* table, bf16* xemb) {
  const long total = (long)SQ * BB * KP;
  for (long id = (long)blockIdx.x * blockDim.x + threadIdx.x; id < total;
       id += (long)gridDim.x * blockDim.x) {
    int k = (int)(id % KP);
    long row = id / KP;
    int b = (int)(row % BB);
    int s = (int)(row / BB);
    int tok = sentence[b * SQ + s];
    float v = (k < EE) ? table[(long)tok * EE + k] : 0.f;
    xemb[id] = f2bf(v);
  }
}

// ---------------- persistent LSTM scan: 2 dirs x 16 batch slices ----------------
__global__ void k_scan(const bf16* __restrict__ whh_pad, const bf16* __restrict__ wih_pad,
                       const float* __restrict__ bias_sum, const bf16* __restrict__ xemb,
                       bf16* __restrict__ hcat) {
  extern __shared__ char smem_raw[];
  bf16*  whh      = (bf16*)smem_raw;               // [GP][KP] bf16
  bf16*  wih      = whh + GP * KP;                 // [GP][KP] bf16
  float* gates    = (float*)(wih + GP * KP);       // [16][GP] f32
  float* c_st     = gates + 16 * GP;               // [16][112] f32
  float* bias_lds = c_st + 16 * 112;               // [GP] f32
  bf16*  h_st     = (bf16*)(bias_lds + GP);        // [16][KP] bf16
  bf16*  x_st     = h_st + 16 * KP;                // 2 x [16][KP] bf16 (double buffer)

  const int tid  = threadIdx.x;
  const int lane = tid & 31;
  const int wave = tid >> 5;
  const int dir  = blockIdx.x >> 4;
  const int b0   = (blockIdx.x & 15) * 16;
  const int TILE = 16 * KP;  // elements per x tile

  // stage weights + bias into LDS (dword copies)
  {
    const unsigned* src_h = (const unsigned*)(whh_pad + (long)dir * GP * KP);
    const unsigned* src_i = (const unsigned*)(wih_pad + (long)dir * GP * KP);
    unsigned* dst_h = (unsigned*)whh;
    unsigned* dst_i = (unsigned*)wih;
    const int nd = GP * KP / 2;
    for (int i = tid; i < nd; i += blockDim.x) { dst_h[i] = src_h[i]; dst_i[i] = src_i[i]; }
  }
  for (int i = tid; i < GP; i += blockDim.x) bias_lds[i] = bias_sum[dir * GP + i];
  for (int i = tid; i < 16 * KP; i += blockDim.x)  h_st[i] = f2bf(0.f);
  for (int i = tid; i < 16 * 112; i += blockDim.x) c_st[i] = 0.f;

  // preload x tile for step 0 into buffer 0
  {
    const int s0 = dir ? (SQ - 1) : 0;
    const char* src = (const char*)(xemb + ((long)s0 * BB + b0) * KP);
#if HAVE_ASYNC_LDS
    async_copy_b128(src + tid * 16, (char*)x_st + tid * 16);
    wait_async0();
#else
    const unsigned* s4 = (const unsigned*)src;
    unsigned* d4 = (unsigned*)x_st;
    for (int i = tid; i < TILE / 2; i += blockDim.x) d4[i] = s4[i];
#endif
  }
  __syncthreads();

  for (int step = 0; step < SQ; ++step) {
    const int s = dir ? (SQ - 1 - step) : step;
    const bf16* xa = x_st + (step & 1) * TILE;
    bf16* xb = x_st + ((step + 1) & 1) * TILE;

    // prefetch next step's x tile into the other buffer (overlaps with WMMA)
    if (step + 1 < SQ) {
      const int sn = dir ? (SQ - 2 - step) : (step + 1);
      const char* src = (const char*)(xemb + ((long)sn * BB + b0) * KP);
#if HAVE_ASYNC_LDS
      async_copy_b128(src + tid * 16, (char*)xb + tid * 16);
#else
      const unsigned* s4 = (const unsigned*)src;
      unsigned* d4 = (unsigned*)xb;
      for (int i = tid; i < TILE / 2; i += blockDim.x) d4[i] = s4[i];
#endif
    }

    // gates = h @ Whh^T + x @ Wih^T + (b_ih+b_hh), gate-padded to 448 cols
    for (int nt = wave; nt < GP / 16; nt += 8) {
      v8f acc = {0.f, 0.f, 0.f, 0.f, 0.f, 0.f, 0.f, 0.f};
#pragma unroll
      for (int kb = 0; kb < 4; ++kb) {
        v16bf a  = load_a_frag(h_st, KP, lane, kb * 32);
        v16bf bm = load_b_frag(whh + nt * 16 * KP, KP, lane, kb * 32);
        acc = __builtin_amdgcn_wmma_f32_16x16x32_bf16(false, a, false, bm,
                                                      (short)0, acc, false, false);
      }
#pragma unroll
      for (int kb = 0; kb < 4; ++kb) {
        v16bf a  = load_a_frag(xa, KP, lane, kb * 32);
        v16bf bm = load_b_frag(wih + nt * 16 * KP, KP, lane, kb * 32);
        acc = __builtin_amdgcn_wmma_f32_16x16x32_bf16(false, a, false, bm,
                                                      (short)0, acc, false, false);
      }
      const int n = lane & 15;
      const int ncol = nt * 16 + n;
      const float bv = bias_lds[ncol];
#pragma unroll
      for (int v = 0; v < 8; ++v) {
        int m = v + 8 * (lane >> 4);
        gates[m * GP + ncol] = acc[v] + bv;
      }
    }
    __syncthreads();

    // activations + state update; write h to global hcat (bf16)
    for (int idx = tid; idx < 16 * 112; idx += blockDim.x) {
      int bi = idx / 112, j = idx % 112;
      float iv = gates[bi * GP + j];
      float fv = gates[bi * GP + 112 + j];
      float gv = gates[bi * GP + 224 + j];
      float ov = gates[bi * GP + 336 + j];
      float ig = 1.f / (1.f + __expf(-iv));
      float fg = 1.f / (1.f + __expf(-fv));
      float gg = tanhf(gv);
      float og = 1.f / (1.f + __expf(-ov));
      float c = fg * c_st[idx] + ig * gg;
      c_st[idx] = c;
      if (j < HH) {
        bf16 hb = f2bf(og * tanhf(c));
        h_st[bi * KP + j] = hb;  // padding lanes stay zero
        hcat[((long)s * BB + b0 + bi) * KC + dir * KP + j] = hb;
      }
    }
#if HAVE_ASYNC_LDS
    wait_async0();  // next x tile resident before anyone reads it next step
#endif
    __syncthreads();
  }
}

// ---------------- emissions GEMM: [S*B][KC] x [NEMI][KC]^T ----------------
__global__ void k_emis(const bf16* __restrict__ hcat, const bf16* __restrict__ woT,
                       const float* __restrict__ b_out, float* __restrict__ emis) {
  const int lane = threadIdx.x & 31;
  const int wave = threadIdx.x >> 5;
  const int tile = blockIdx.x * 8 + wave;   // 16384 tiles total
  const int mt = tile >> 1;
  const int nt = tile & 1;
  v8f acc = {0.f, 0.f, 0.f, 0.f, 0.f, 0.f, 0.f, 0.f};
#pragma unroll
  for (int kb = 0; kb < 8; ++kb) {
    v16bf a  = load_a_frag(hcat + (long)mt * 16 * KC, KC, lane, kb * 32);
    v16bf bm = load_b_frag(woT + nt * 16 * KC, KC, lane, kb * 32);
    acc = __builtin_amdgcn_wmma_f32_16x16x32_bf16(false, a, false, bm,
                                                  (short)0, acc, false, false);
  }
  const int n = lane & 15;
  const int ncol = nt * 16 + n;
  const float bv = (ncol < TT) ? b_out[ncol] : 0.f;
#pragma unroll
  for (int v = 0; v < 8; ++v) {
    int m = v + 8 * (lane >> 4);
    long row = (long)mt * 16 + m;
    emis[row * NEMI + ncol] = acc[v] + bv;
  }
}

// ---------------- CRF logZ: one wave32 per batch row ----------------
__global__ void k_logz(const float* __restrict__ emis, const unsigned char* __restrict__ mask,
                       const float* __restrict__ start_t, const float* __restrict__ end_t,
                       const float* __restrict__ trans, float* __restrict__ logZ) {
  const int lane = threadIdx.x & 31;
  const int b = (blockIdx.x * blockDim.x + threadIdx.x) >> 5;
  const float NEG = -3.0e38f;
  float tr[TT];
#pragma unroll
  for (int i = 0; i < TT; ++i) tr[i] = (lane < TT) ? trans[i * TT + lane] : NEG;
  float score = (lane < TT) ? (start_t[lane] + emis[(long)b * NEMI + lane]) : NEG;
  for (int s = 1; s < SQ; ++s) {
    float v[TT];
    float m = NEG;
#pragma unroll
    for (int i = 0; i < TT; ++i) {
      float si = __shfl(score, i, 32);
      v[i] = si + tr[i];
      m = fmaxf(m, v[i]);
    }
    float sum = 0.f;
#pragma unroll
    for (int i = 0; i < TT; ++i) sum += __expf(v[i] - m);
    float nxt = m + __logf(sum) + emis[((long)s * BB + b) * NEMI + lane];
    if (lane < TT && mask[b * SQ + s]) score = nxt;
  }
  float fs = (lane < TT) ? (score + end_t[lane]) : NEG;
  float mx = fs;
  for (int off = 16; off > 0; off >>= 1) mx = fmaxf(mx, __shfl_xor(mx, off, 32));
  float e = (lane < TT) ? __expf(fs - mx) : 0.f;
  for (int off = 16; off > 0; off >>= 1) e += __shfl_xor(e, off, 32);
  if (lane == 0) logZ[b] = mx + __logf(e);
}

// ---------------- CRF numerator: one thread per batch row ----------------
__global__ void k_num(const float* __restrict__ emis, const int* __restrict__ tags,
                      const unsigned char* __restrict__ mask,
                      const float* __restrict__ start_t, const float* __restrict__ end_t,
                      const float* __restrict__ trans, float* __restrict__ num) {
  int b = blockIdx.x * blockDim.x + threadIdx.x;
  if (b >= BB) return;
  int t0 = mask[b * SQ] ? tags[b * SQ] : 0;
  float sc = start_t[t0] + emis[(long)b * NEMI + t0];
  int prev = t0, last = t0;
  for (int s = 1; s < SQ; ++s) {
    int mk = mask[b * SQ + s] ? 1 : 0;
    int tc = mk ? tags[b * SQ + s] : 0;
    float contrib = trans[prev * TT + tc] + emis[((long)s * BB + b) * NEMI + tc];
    if (mk) { sc += contrib; last = tc; }
    prev = tc;
  }
  num[b] = sc + end_t[last];
}

__global__ void k_reduce(const float* logZ, const float* num, float* out) {
  __shared__ float sh[256];
  int t = threadIdx.x;
  sh[t] = logZ[t] - num[t];
  __syncthreads();
  for (int off = 128; off > 0; off >>= 1) {
    if (t < off) sh[t] += sh[t + off];
    __syncthreads();
  }
  if (t == 0) out[0] = sh[0];
}

extern "C" void kernel_launch(void* const* d_in, const int* in_sizes, int n_in,
                              void* d_out, int out_size, void* d_ws, size_t ws_size,
                              hipStream_t stream) {
  (void)in_sizes; (void)n_in; (void)out_size; (void)ws_size;
  const int*   sentence = (const int*)d_in[0];
  const int*   tags     = (const int*)d_in[1];
  const unsigned char* mask = (const unsigned char*)d_in[2];
  const float* table    = (const float*)d_in[3];
  const float* w_ih_f   = (const float*)d_in[4];
  const float* w_hh_f   = (const float*)d_in[5];
  const float* b_ih_f   = (const float*)d_in[6];
  const float* b_hh_f   = (const float*)d_in[7];
  const float* w_ih_b   = (const float*)d_in[8];
  const float* w_hh_b   = (const float*)d_in[9];
  const float* b_ih_b   = (const float*)d_in[10];
  const float* b_hh_b   = (const float*)d_in[11];
  const float* w_out    = (const float*)d_in[12];
  const float* b_out    = (const float*)d_in[13];
  const float* start_t  = (const float*)d_in[14];
  const float* end_t    = (const float*)d_in[15];
  const float* trans    = (const float*)d_in[16];

  char* ws = (char*)d_ws;
  size_t off = 0;
  auto alloc = [&](size_t bytes) -> void* {
    void* p = ws + off;
    off = (off + bytes + 255) & ~(size_t)255;
    return p;
  };
  bf16*  whh_pad = (bf16*)alloc(2ull * GP * KP * sizeof(bf16));
  bf16*  wih_pad = (bf16*)alloc(2ull * GP * KP * sizeof(bf16));
  float* bias    = (float*)alloc(2ull * GP * sizeof(float));
  bf16*  woT     = (bf16*)alloc((size_t)NEMI * KC * sizeof(bf16));
  bf16*  xemb    = (bf16*)alloc((size_t)SQ * BB * KP * sizeof(bf16));
  bf16*  hcat    = (bf16*)alloc((size_t)SQ * BB * KC * sizeof(bf16));
  float* emis    = (float*)alloc((size_t)SQ * BB * NEMI * sizeof(float));
  float* logZ    = (float*)alloc((size_t)BB * sizeof(float));
  float* numv    = (float*)alloc((size_t)BB * sizeof(float));

  k_prep_wpad<<<256, 256, 0, stream>>>(w_ih_f, w_hh_f, w_ih_b, w_hh_b, whh_pad, wih_pad);
  k_prep_misc<<<64, 256, 0, stream>>>(b_ih_f, b_hh_f, b_ih_b, b_hh_b, w_out, bias, woT);
  k_embed<<<4096, 256, 0, stream>>>(sentence, table, xemb);
  (void)hipMemsetAsync(hcat, 0, (size_t)SQ * BB * KC * sizeof(bf16), stream);

  const size_t smem = (size_t)(2 * GP * KP * sizeof(bf16))   // weights
                    + (size_t)(16 * GP * sizeof(float))      // gates
                    + (size_t)(16 * 112 * sizeof(float))     // c
                    + (size_t)(GP * sizeof(float))           // bias
                    + (size_t)(16 * KP * sizeof(bf16))       // h
                    + (size_t)(2 * 16 * KP * sizeof(bf16));  // x double buffer
  (void)hipFuncSetAttribute(reinterpret_cast<const void*>(k_scan),
                            hipFuncAttributeMaxDynamicSharedMemorySize, (int)smem);
  k_scan<<<32, 256, smem, stream>>>(whh_pad, wih_pad, bias, xemb, hcat);

  k_emis<<<2048, 256, 0, stream>>>(hcat, woT, b_out, emis);
  k_logz<<<32, 256, 0, stream>>>(emis, mask, start_t, end_t, trans, logZ);
  k_num<<<1, 256, 0, stream>>>(emis, tags, mask, start_t, end_t, trans, numv);
  k_reduce<<<1, 256, 0, stream>>>(logZ, numv, (float*)d_out);
}
